// Pctes_58205396796053
// MI455X (gfx1250) — matrix operand order repaired and checked by reading, not compiled, not run on hardware
//
#include <hip/hip_runtime.h>
#include <hip/hip_bf16.h>
#include <math.h>

typedef __attribute__((ext_vector_type(16))) _Float16 v16h;
typedef __attribute__((ext_vector_type(8)))  float    v8f;

#define DEV static __device__ __forceinline__

// ---------------------------------------------------------------------------
// WMMA helpers (gfx1250, wave32): D(16x16 f32) = A(16x32 f16) * B(32x16 f16) + C
// Fragment layouts as loaded here (A/B in contiguous K order):
//   A: lane L holds row M=L%16; element e -> K = e + 16*(L>=16)
//   B: lane L holds col N=L%16; element e -> K = e + 16*(L>=16)
//   D: lane L holds col N=L%16; element r -> M = r + 8*(L>=16)
// ---------------------------------------------------------------------------
DEV v8f wmma_f16f32(v16h a, v16h b, v8f c) {
  return __builtin_amdgcn_wmma_f32_16x16x32_f16(false, a, false, b, (short)0, c,
                                                false, false);
}

DEV v16h load_a_f16(const _Float16* A, int lda, int m0, int k0, int lane) {
  const _Float16* p = A + (size_t)(m0 + (lane & 15)) * lda + k0;
  int kb = (lane & 16) ? 16 : 0;
  v16h a;
#pragma unroll
  for (int e = 0; e < 16; ++e) { int k = e + kb; a[e] = p[k]; }
  return a;
}

DEV v16h load_a_f32(const float* A, int lda, int m0, int k0, int lane) {
  const float* p = A + (size_t)(m0 + (lane & 15)) * lda + k0;
  int kb = (lane & 16) ? 16 : 0;
  v16h a;
#pragma unroll
  for (int e = 0; e < 16; ++e) { int k = e + kb; a[e] = (_Float16)p[k]; }
  return a;
}

// B stored row-major [N][K] (i.e. C = A * B^T, the "weight[o][c]" case)
DEV v16h load_b_nt(const _Float16* Bm, int ldb, int n0, int k0, int lane) {
  const _Float16* p = Bm + (size_t)(n0 + (lane & 15)) * ldb + k0 + ((lane & 16) ? 16 : 0);
  v16h b;
#pragma unroll
  for (int e = 0; e < 16; ++e) b[e] = p[e];
  return b;
}

// B stored row-major [K][N] (plain C = A * B)
DEV v16h load_b_nn(const _Float16* Bm, int ldb, int n0, int k0, int lane) {
  const _Float16* p = Bm + (size_t)(k0 + ((lane & 16) ? 16 : 0)) * ldb + n0 + (lane & 15);
  v16h b;
#pragma unroll
  for (int e = 0; e < 16; ++e) b[e] = p[(size_t)e * ldb];
  return b;
}

// ---------------------------------------------------------------------------
// Generic one-wave-per-16x16-tile WMMA GEMM with epilogues.
// grid = (ceil(N/16), M/16, batch), block = 32 (one wave).
// ---------------------------------------------------------------------------
enum { EPI_NONE = 0, EPI_BIAS, EPI_BN_RELU, EPI_BN_LRELU, EPI_BIAS_BN_LRELU,
       EPI_SUBFROM, EPI_ACCUM, EPI_T_RES };

template <int EPI, int AF32, int OUTF16, int BNN>
__global__ __launch_bounds__(32)
void gemm16(const void* __restrict__ A, int lda, size_t aBS,
            const _Float16* __restrict__ Bw, int ldb, size_t bBS,
            void* __restrict__ Out, int ldo, size_t oBS,
            const float* __restrict__ scale, const float* __restrict__ shift,
            const float* __restrict__ bias,
            const _Float16* __restrict__ Res, int ldr, size_t rBS,
            int Ncols, int Ktot) {
  int lane = threadIdx.x;
  int n0 = blockIdx.x * 16;
  int m0 = blockIdx.y * 16;
  int b  = blockIdx.z;
  const _Float16* Ah = (const _Float16*)A + (AF32 ? 0 : (size_t)b * aBS);
  const float*    Af = (const float*)A + (AF32 ? (size_t)b * aBS : 0);
  const _Float16* Bp = Bw + (size_t)b * bBS;
  v8f acc = {0.f, 0.f, 0.f, 0.f, 0.f, 0.f, 0.f, 0.f};
  for (int k0 = 0; k0 < Ktot; k0 += 32) {
    v16h a;
    if (AF32) a = load_a_f32(Af, lda, m0, k0, lane);
    else      a = load_a_f16(Ah, lda, m0, k0, lane);
    v16h bf;
    if (BNN) bf = load_b_nn(Bp, ldb, n0, k0, lane);
    else     bf = load_b_nt(Bp, ldb, n0, k0, lane);
    acc = wmma_f16f32(a, bf, acc);
  }
  int col = n0 + (lane & 15);
  if (col >= Ncols) return;
  float sc = 1.f, sh = 0.f, bi = 0.f;
  if (EPI == EPI_BN_RELU || EPI == EPI_BN_LRELU || EPI == EPI_BIAS_BN_LRELU ||
      EPI == EPI_T_RES) { sc = scale[col]; sh = shift[col]; }
  if (EPI == EPI_BIAS || EPI == EPI_BIAS_BN_LRELU || EPI == EPI_T_RES) bi = bias[col];
  const _Float16* Rp =
      (EPI == EPI_SUBFROM || EPI == EPI_T_RES) ? Res + (size_t)b * rBS : nullptr;
#pragma unroll
  for (int r = 0; r < 8; ++r) {
    int row = m0 + r + ((lane & 16) ? 8 : 0);
    float v = acc[r];
    if (EPI == EPI_BIAS) v += bi;
    if (EPI == EPI_BN_RELU)  { v = v * sc + sh; v = v > 0.f ? v : 0.f; }
    if (EPI == EPI_BN_LRELU) { v = v * sc + sh; v = v > 0.f ? v : 0.2f * v; }
    if (EPI == EPI_BIAS_BN_LRELU) { v = (v + bi) * sc + sh; v = v > 0.f ? v : 0.2f * v; }
    if (EPI == EPI_SUBFROM) v = (float)Rp[(size_t)row * ldr + col] - v;
    if (EPI == EPI_T_RES) {
      v = (v + bi) * sc + sh; v = v > 0.f ? v : 0.f;
      v += (float)Rp[(size_t)row * ldr + col];
    }
    if (OUTF16) {
      ((_Float16*)Out)[(size_t)b * oBS + (size_t)row * ldo + col] = (_Float16)v;
    } else {
      float* op = (float*)Out + (size_t)b * oBS + (size_t)row * ldo + col;
      if (EPI == EPI_ACCUM) *op = *op + v; else *op = v;
    }
  }
}

// ---------------------------------------------------------------------------
// RFF embedding: x [B,3,N] -> xyz [B,N,3], feat [B,N,64] = [sin|cos](2*pi*xyz@rb^T)
// ---------------------------------------------------------------------------
__global__ void rff_kernel(const float* __restrict__ x, const float* __restrict__ rb,
                           float* __restrict__ xyz, float* __restrict__ feat) {
  int i = blockIdx.x * 256 + threadIdx.x;
  if (i >= 16 * 1024) return;
  int b = i >> 10, n = i & 1023;
  float X = x[((size_t)b * 3 + 0) * 1024 + n];
  float Y = x[((size_t)b * 3 + 1) * 1024 + n];
  float Z = x[((size_t)b * 3 + 2) * 1024 + n];
  xyz[(size_t)i * 3 + 0] = X; xyz[(size_t)i * 3 + 1] = Y; xyz[(size_t)i * 3 + 2] = Z;
  const float TP = 6.283185307179586f;
#pragma unroll 4
  for (int f = 0; f < 32; ++f) {
    float p = TP * (X * rb[f * 3] + Y * rb[f * 3 + 1] + Z * rb[f * 3 + 2]);
    feat[(size_t)i * 64 + f]      = sinf(p);
    feat[(size_t)i * 64 + 32 + f] = cosf(p);
  }
}

// ---------------------------------------------------------------------------
// Farthest point sampling: one 256-thread block per batch; deterministic start 0.
// Matches jax scan semantics: emit `far` then update distances.
// ---------------------------------------------------------------------------
template <int SRCN, int S>
__global__ __launch_bounds__(256)
void fps_kernel(const float* __restrict__ xyz, int* __restrict__ fidx,
                float* __restrict__ nxyz) {
  constexpr int PPT = SRCN / 256;
  __shared__ float sv[256];
  __shared__ int   si[256];
  __shared__ int   sFar;
  int b = blockIdx.x, tid = threadIdx.x;
  const float* xb = xyz + (size_t)b * SRCN * 3;
  float px[PPT], py[PPT], pz[PPT], dd[PPT];
#pragma unroll
  for (int i = 0; i < PPT; ++i) {
    int p = tid + i * 256;
    px[i] = xb[p * 3]; py[i] = xb[p * 3 + 1]; pz[i] = xb[p * 3 + 2];
    dd[i] = 1e10f;
  }
  if (tid == 0) sFar = 0;
  __syncthreads();
  for (int it = 0; it < S; ++it) {
    int far = sFar;
    float cx = xb[far * 3], cy = xb[far * 3 + 1], cz = xb[far * 3 + 2];
    if (tid == 0) {
      fidx[b * S + it] = far;
      nxyz[((size_t)b * S + it) * 3 + 0] = cx;
      nxyz[((size_t)b * S + it) * 3 + 1] = cy;
      nxyz[((size_t)b * S + it) * 3 + 2] = cz;
    }
    float bv = -1.f; int bi = 0;
#pragma unroll
    for (int i = 0; i < PPT; ++i) {
      float dx = px[i] - cx, dy = py[i] - cy, dz = pz[i] - cz;
      float d = dx * dx + dy * dy + dz * dz;
      dd[i] = fminf(dd[i], d);
      if (dd[i] > bv) { bv = dd[i]; bi = tid + i * 256; }
    }
    sv[tid] = bv; si[tid] = bi;
    __syncthreads();
    for (int off = 128; off > 0; off >>= 1) {
      if (tid < off) {
        float v2 = sv[tid + off]; int i2 = si[tid + off];
        if (v2 > sv[tid] || (v2 == sv[tid] && i2 < si[tid])) { sv[tid] = v2; si[tid] = i2; }
      }
      __syncthreads();
    }
    if (tid == 0) sFar = si[0];
    __syncthreads();
  }
}

// ---------------------------------------------------------------------------
// Brute-force kNN (k=32): one block per (s,b); 32 argmin passes over LDS dists.
// Lower-index tie-break matches stable top_k.
// ---------------------------------------------------------------------------
template <int SRCN>
__global__ __launch_bounds__(256)
void knn_kernel(const float* __restrict__ src, const float* __restrict__ nxyz,
                int* __restrict__ knnOut) {
  __shared__ float d[SRCN];
  __shared__ float sv[256];
  __shared__ int   si[256];
  int s = blockIdx.x, b = blockIdx.y, S = gridDim.x, tid = threadIdx.x;
  const float* sb = src + (size_t)b * SRCN * 3;
  float cx = nxyz[((size_t)b * S + s) * 3 + 0];
  float cy = nxyz[((size_t)b * S + s) * 3 + 1];
  float cz = nxyz[((size_t)b * S + s) * 3 + 2];
  for (int p = tid; p < SRCN; p += 256) {
    float dx = sb[p * 3] - cx, dy = sb[p * 3 + 1] - cy, dz = sb[p * 3 + 2] - cz;
    d[p] = dx * dx + dy * dy + dz * dz;
  }
  __syncthreads();
  for (int k = 0; k < 32; ++k) {
    float bv = 3.3e38f; int bi = 0;
    for (int p = tid; p < SRCN; p += 256) {
      float v = d[p];
      if (v < bv) { bv = v; bi = p; }
    }
    sv[tid] = bv; si[tid] = bi;
    __syncthreads();
    for (int off = 128; off > 0; off >>= 1) {
      if (tid < off) {
        float v2 = sv[tid + off]; int i2 = si[tid + off];
        if (v2 < sv[tid] || (v2 == sv[tid] && i2 < si[tid])) { sv[tid] = v2; si[tid] = i2; }
      }
      __syncthreads();
    }
    if (tid == 0) { knnOut[((size_t)b * S + s) * 32 + k] = si[0]; d[si[0]] = 3.3e38f; }
    __syncthreads();
  }
}

// ---------------------------------------------------------------------------
// Fused sample_and_group + local_op: per (s,b) block gathers 32 neighbors
// [rel | ctr] into LDS (f16), two WMMA GEMMs with BN+ReLU, max over neighbors.
// ---------------------------------------------------------------------------
template <int DIN, int DOUT, int OUTF16>
__global__ __launch_bounds__(256)
void local_op_kernel(const float* __restrict__ pts, int SRCN,
                     const int* __restrict__ fidx, const int* __restrict__ knnIdx,
                     const _Float16* __restrict__ w1, const _Float16* __restrict__ w2,
                     const float* __restrict__ s1, const float* __restrict__ t1,
                     const float* __restrict__ s2, const float* __restrict__ t2,
                     void* __restrict__ out) {
  constexpr int DIN2 = 2 * DIN;
  __shared__ _Float16 Xs[32 * DIN2];
  __shared__ _Float16 Ys[32 * DOUT];
  __shared__ float    Zs[32 * DOUT];
  __shared__ int      nb[32];
  int s = blockIdx.x, b = blockIdx.y, S = gridDim.x, tid = threadIdx.x;
  if (tid < 32) nb[tid] = knnIdx[((size_t)b * S + s) * 32 + tid];
  __syncthreads();
  const float* pB = pts + (size_t)b * SRCN * DIN;
  const float* cp = pB + (size_t)fidx[b * S + s] * DIN;
  for (int i = tid; i < 32 * DIN; i += 256) {
    int j = i / DIN, dch = i - j * DIN;
    float cd = cp[dch];
    Xs[j * DIN2 + dch]       = (_Float16)(pB[(size_t)nb[j] * DIN + dch] - cd);
    Xs[j * DIN2 + DIN + dch] = (_Float16)cd;
  }
  __syncthreads();
  int wv = tid >> 5, lane = tid & 31;
  // GEMM1: Y = relu(bn1(X * W1^T)), X 32xDIN2, W1 [DOUT][DIN2]
  for (int t = wv; t < 2 * (DOUT / 16); t += 8) {
    int mi = t & 1, ni = t >> 1;
    v8f acc = {0.f, 0.f, 0.f, 0.f, 0.f, 0.f, 0.f, 0.f};
    for (int k0 = 0; k0 < DIN2; k0 += 32) {
      v16h a  = load_a_f16(Xs, DIN2, mi * 16, k0, lane);
      v16h bf = load_b_nt(w1, DIN2, ni * 16, k0, lane);
      acc = wmma_f16f32(a, bf, acc);
    }
    int col = ni * 16 + (lane & 15);
    float sc = s1[col], sh = t1[col];
#pragma unroll
    for (int r = 0; r < 8; ++r) {
      int row = mi * 16 + r + ((lane & 16) ? 8 : 0);
      float v = acc[r] * sc + sh; v = v > 0.f ? v : 0.f;
      Ys[row * DOUT + col] = (_Float16)v;
    }
  }
  __syncthreads();
  // GEMM2: Z = relu(bn2(Y * W2^T)), W2 [DOUT][DOUT]
  for (int t = wv; t < 2 * (DOUT / 16); t += 8) {
    int mi = t & 1, ni = t >> 1;
    v8f acc = {0.f, 0.f, 0.f, 0.f, 0.f, 0.f, 0.f, 0.f};
    for (int k0 = 0; k0 < DOUT; k0 += 32) {
      v16h a  = load_a_f16(Ys, DOUT, mi * 16, k0, lane);
      v16h bf = load_b_nt(w2, DOUT, ni * 16, k0, lane);
      acc = wmma_f16f32(a, bf, acc);
    }
    int col = ni * 16 + (lane & 15);
    float sc = s2[col], sh = t2[col];
#pragma unroll
    for (int r = 0; r < 8; ++r) {
      int row = mi * 16 + r + ((lane & 16) ? 8 : 0);
      float v = acc[r] * sc + sh; v = v > 0.f ? v : 0.f;
      Zs[row * DOUT + col] = v;
    }
  }
  __syncthreads();
  size_t ob = ((size_t)b * S + s) * DOUT;
  for (int ch = tid; ch < DOUT; ch += 256) {
    float mx = Zs[ch];
#pragma unroll 4
    for (int j = 1; j < 32; ++j) mx = fmaxf(mx, Zs[j * DOUT + ch]);
    if (OUTF16) ((_Float16*)out)[ob + ch] = (_Float16)mx;
    else        ((float*)out)[ob + ch] = mx;
  }
}

// ---------------------------------------------------------------------------
// softmax over last axis of E[b][n][:256], written TRANSPOSED: AT[b][m][n]
// ---------------------------------------------------------------------------
__global__ __launch_bounds__(256)
void softmaxT_kernel(const float* __restrict__ E, float* __restrict__ AT) {
  __shared__ float red[256];
  int n = blockIdx.x, b = blockIdx.y, m = threadIdx.x;
  float v = E[((size_t)b * 256 + n) * 256 + m];
  red[m] = v; __syncthreads();
  for (int off = 128; off > 0; off >>= 1) {
    if (m < off) red[m] = fmaxf(red[m], red[m + off]);
    __syncthreads();
  }
  float mx = red[0];
  __syncthreads();
  float e = expf(v - mx);
  red[m] = e; __syncthreads();
  for (int off = 128; off > 0; off >>= 1) {
    if (m < off) red[m] += red[m + off];
    __syncthreads();
  }
  AT[((size_t)b * 256 + m) * 256 + n] = e / red[0];
}

// L1 renorm over queries n: AT[b][m][n] /= (1e-9 + sum_n AT[b][m][n])
__global__ __launch_bounds__(256)
void renorm_kernel(float* __restrict__ AT) {
  __shared__ float red[256];
  int m = blockIdx.x, b = blockIdx.y, n = threadIdx.x;
  size_t off = ((size_t)b * 256 + m) * 256 + n;
  float v = AT[off];
  red[n] = v; __syncthreads();
  for (int o = 128; o > 0; o >>= 1) {
    if (n < o) red[n] += red[n + o];
    __syncthreads();
  }
  AT[off] = v / (1e-9f + red[0]);
}

// mean over positions of lrelu(bn(F)): pooled[b][o] = mean_n lrelu(bn(F[b][n][o]))
__global__ void pool_kernel(const float* __restrict__ F, const float* __restrict__ sc,
                            const float* __restrict__ sh, float* __restrict__ pooled) {
  int i = blockIdx.x * 256 + threadIdx.x;
  if (i >= 16 * 1024) return;
  int b = i >> 10, o = i & 1023;
  const float* fp = F + (size_t)b * 256 * 1024 + o;
  float s = sc[o], t = sh[o], acc = 0.f;
  for (int n = 0; n < 256; ++n) {
    float v = fp[(size_t)n * 1024] * s + t;
    acc += (v > 0.f ? v : 0.2f * v);
  }
  pooled[i] = acc * (1.f / 256.f);
}

__global__ void bn_prep_kernel(const float* __restrict__ g, const float* __restrict__ b,
                               const float* __restrict__ m, const float* __restrict__ v,
                               float* __restrict__ sc, float* __restrict__ sh, int n) {
  int i = blockIdx.x * 256 + threadIdx.x;
  if (i < n) {
    float s = g[i] * rsqrtf(v[i] + 1e-5f);
    sc[i] = s; sh[i] = b[i] - m[i] * s;
  }
}

__global__ void f16_convert_kernel(const float* __restrict__ src, _Float16* __restrict__ dst,
                                   int n, int ntot) {
  int i = blockIdx.x * 256 + threadIdx.x;
  if (i < ntot) dst[i] = (i < n) ? (_Float16)src[i] : (_Float16)0.f;
}

// ---------------------------------------------------------------------------
// Host orchestration
// ---------------------------------------------------------------------------
extern "C" void kernel_launch(void* const* d_in, const int* in_sizes, int n_in,
                              void* d_out, int out_size, void* d_ws, size_t ws_size,
                              hipStream_t stream) {
  (void)in_sizes; (void)n_in; (void)out_size; (void)ws_size;
  const int B = 16, N = 1024, S0 = 512, S1 = 256;
  char* base = (char*)d_ws;
  size_t off = 0;
  auto alloc = [&](size_t bytes) -> void* {
    off = (off + 255) & ~(size_t)255;
    void* p = base + off; off += bytes; return p;
  };
  // activations / geometry
  float* xyz   = (float*)alloc((size_t)B * N * 3 * 4);
  float* feat  = (float*)alloc((size_t)B * N * 64 * 4);
  int*   fidx0 = (int*)alloc((size_t)B * S0 * 4);
  float* nxyz0 = (float*)alloc((size_t)B * S0 * 3 * 4);
  int*   knn0  = (int*)alloc((size_t)B * S0 * 32 * 4);
  float* f0    = (float*)alloc((size_t)B * S0 * 128 * 4);
  int*   fidx1 = (int*)alloc((size_t)B * S1 * 4);
  float* nxyz1 = (float*)alloc((size_t)B * S1 * 3 * 4);
  int*   knn1  = (int*)alloc((size_t)B * S1 * 32 * 4);
  _Float16* f1h = (_Float16*)alloc((size_t)B * S1 * 256 * 2);
  // f16 weights
  _Float16* lo0w1h = (_Float16*)alloc(16384 * 2);
  _Float16* lo0w2h = (_Float16*)alloc(16384 * 2);
  _Float16* lo1w1h = (_Float16*)alloc(65536 * 2);
  _Float16* lo1w2h = (_Float16*)alloc(65536 * 2);
  _Float16* ptw1h  = (_Float16*)alloc(65536 * 2);
  _Float16* ptw2h  = (_Float16*)alloc(65536 * 2);
  _Float16* wqh[4], *wvh[4], *wth[4];
  for (int i = 0; i < 4; ++i) {
    wqh[i] = (_Float16*)alloc(16384 * 2);
    wvh[i] = (_Float16*)alloc(65536 * 2);
    wth[i] = (_Float16*)alloc(65536 * 2);
  }
  _Float16* fusewh = (_Float16*)alloc((size_t)1310720 * 2);
  _Float16* l1h = (_Float16*)alloc((size_t)524288 * 2);
  _Float16* l2h = (_Float16*)alloc((size_t)131072 * 2);
  _Float16* l3h = (_Float16*)alloc((size_t)48 * 256 * 2);  // padded 40->48 rows
  // folded bn scale/shift
  auto fpair = [&](int n, float** s, float** t) {
    *s = (float*)alloc(n * 4); *t = (float*)alloc(n * 4);
  };
  float *lo0s1,*lo0t1,*lo0s2,*lo0t2,*lo1s1,*lo1t1,*lo1s2,*lo1t2;
  float *pts1,*ptt1,*pts2,*ptt2,*fuseS,*fuseT,*bn6S,*bn6T,*bn7S,*bn7T;
  float *saS[4],*saT[4];
  fpair(128,&lo0s1,&lo0t1); fpair(128,&lo0s2,&lo0t2);
  fpair(256,&lo1s1,&lo1t1); fpair(256,&lo1s2,&lo1t2);
  fpair(256,&pts1,&ptt1);   fpair(256,&pts2,&ptt2);
  for (int i = 0; i < 4; ++i) fpair(256,&saS[i],&saT[i]);
  fpair(1024,&fuseS,&fuseT); fpair(512,&bn6S,&bn6T); fpair(256,&bn7S,&bn7T);
  // attention / head buffers
  _Float16* yh  = (_Float16*)alloc((size_t)B * 256 * 256 * 2);
  _Float16* x0h = (_Float16*)alloc((size_t)B * 256 * 256 * 2);
  _Float16* x1h = (_Float16*)alloc((size_t)B * 256 * 256 * 2);
  _Float16* x2h = (_Float16*)alloc((size_t)B * 256 * 256 * 2);
  _Float16* x3h = (_Float16*)alloc((size_t)B * 256 * 256 * 2);
  _Float16* x4h = (_Float16*)alloc((size_t)B * 256 * 256 * 2);
  _Float16* Qh  = (_Float16*)alloc((size_t)B * 256 * 64 * 2);
  _Float16* Vh  = (_Float16*)alloc((size_t)B * 256 * 256 * 2);
  _Float16* Uh  = (_Float16*)alloc((size_t)B * 256 * 256 * 2);
  float* E  = (float*)alloc((size_t)B * 256 * 256 * 4);
  float* AT = (float*)alloc((size_t)B * 256 * 256 * 4);
  float* F  = (float*)alloc((size_t)B * 256 * 1024 * 4);
  float* pooled = (float*)alloc((size_t)B * 1024 * 4);
  float* h1b = (float*)alloc((size_t)B * 512 * 4);
  float* h2b = (float*)alloc((size_t)B * 256 * 4);

  // ---- input index map (recursive insertion order of setup_inputs dict) ----
  // 0:x 1:rff_b 2:lo0.w1 3-6:lo0.bn1 7:lo0.w2 8-11:lo0.bn2 12:lo1.w1 13-16:lo1.bn1
  // 17:lo1.w2 18-21:lo1.bn2 22:pt.w1 23-26:pt.bn1 27:pt.w2 28-31:pt.bn2
  // sa_i @ {32,41,50,59}: wq,wv,bv,wt,bt,bn{g,b,m,v}
  // 68:fuse_w 69-72:fuse_bn 73:l1 74-77:bn6 78:l2 79:b2 80-83:bn7 84:l3 85:b3
  const int saBase[4] = {32, 41, 50, 59};
  auto F32 = [&](int i) { return (const float*)d_in[i]; };

  auto conv = [&](int idx, _Float16* dst, int n, int ntot) {
    f16_convert_kernel<<<(ntot + 255) / 256, 256, 0, stream>>>(F32(idx), dst, n, ntot);
  };
  conv(2, lo0w1h, 16384, 16384);  conv(7, lo0w2h, 16384, 16384);
  conv(12, lo1w1h, 65536, 65536); conv(17, lo1w2h, 65536, 65536);
  conv(22, ptw1h, 65536, 65536);  conv(27, ptw2h, 65536, 65536);
  for (int i = 0; i < 4; ++i) {
    conv(saBase[i] + 0, wqh[i], 16384, 16384);
    conv(saBase[i] + 1, wvh[i], 65536, 65536);
    conv(saBase[i] + 3, wth[i], 65536, 65536);
  }
  conv(68, fusewh, 1310720, 1310720);
  conv(73, l1h, 524288, 524288);
  conv(78, l2h, 131072, 131072);
  conv(84, l3h, 10240, 12288);

  auto bnp = [&](int gi, float* s, float* t, int n) {
    bn_prep_kernel<<<(n + 255) / 256, 256, 0, stream>>>(F32(gi), F32(gi + 1), F32(gi + 2),
                                                        F32(gi + 3), s, t, n);
  };
  bnp(3, lo0s1, lo0t1, 128); bnp(8, lo0s2, lo0t2, 128);
  bnp(13, lo1s1, lo1t1, 256); bnp(18, lo1s2, lo1t2, 256);
  bnp(23, pts1, ptt1, 256);  bnp(28, pts2, ptt2, 256);
  for (int i = 0; i < 4; ++i) bnp(saBase[i] + 5, saS[i], saT[i], 256);
  bnp(69, fuseS, fuseT, 1024); bnp(74, bn6S, bn6T, 512); bnp(80, bn7S, bn7T, 256);

  // ---- pipeline ----
  rff_kernel<<<(B * N + 255) / 256, 256, 0, stream>>>(F32(0), F32(1), xyz, feat);
  fps_kernel<1024, 512><<<16, 256, 0, stream>>>(xyz, fidx0, nxyz0);
  knn_kernel<1024><<<dim3(512, 16), 256, 0, stream>>>(xyz, nxyz0, knn0);
  local_op_kernel<64, 128, 0><<<dim3(512, 16), 256, 0, stream>>>(
      feat, 1024, fidx0, knn0, lo0w1h, lo0w2h, lo0s1, lo0t1, lo0s2, lo0t2, f0);
  fps_kernel<512, 256><<<16, 256, 0, stream>>>(nxyz0, fidx1, nxyz1);
  knn_kernel<512><<<dim3(256, 16), 256, 0, stream>>>(nxyz0, nxyz1, knn1);
  local_op_kernel<128, 256, 1><<<dim3(256, 16), 256, 0, stream>>>(
      f0, 512, fidx1, knn1, lo1w1h, lo1w2h, lo1s1, lo1t1, lo1s2, lo1t2, f1h);

  const size_t A256 = 65536, Q64 = 16384;
  dim3 g256(16, 16, 16);
  // pt conv1 / conv2
  gemm16<EPI_BN_RELU, 0, 1, 0><<<g256, 32, 0, stream>>>(
      f1h, 256, A256, ptw1h, 256, 0, yh, 256, A256, pts1, ptt1, nullptr,
      nullptr, 0, 0, 256, 256);
  gemm16<EPI_BN_RELU, 0, 1, 0><<<g256, 32, 0, stream>>>(
      yh, 256, A256, ptw2h, 256, 0, x0h, 256, A256, pts2, ptt2, nullptr,
      nullptr, 0, 0, 256, 256);

  _Float16* cur = x0h;
  _Float16* outs[4] = {x1h, x2h, x3h, x4h};
  for (int i = 0; i < 4; ++i) {
    const float* bv = F32(saBase[i] + 2);
    const float* bt = F32(saBase[i] + 4);
    // Q = x * Wq^T  [B,256,64]  (k projection is tied -> energy = Q Q^T)
    gemm16<EPI_NONE, 0, 1, 0><<<dim3(4, 16, 16), 32, 0, stream>>>(
        cur, 256, A256, wqh[i], 256, 0, Qh, 64, Q64, nullptr, nullptr, nullptr,
        nullptr, 0, 0, 64, 256);
    // V = x * Wv^T + bv
    gemm16<EPI_BIAS, 0, 1, 0><<<g256, 32, 0, stream>>>(
        cur, 256, A256, wvh[i], 256, 0, Vh, 256, A256, nullptr, nullptr, bv,
        nullptr, 0, 0, 256, 256);
    // E[n][m] = Q[n] . Q[m]
    gemm16<EPI_NONE, 0, 0, 0><<<g256, 32, 0, stream>>>(
        Qh, 64, Q64, Qh, 64, Q64, E, 256, A256, nullptr, nullptr, nullptr,
        nullptr, 0, 0, 256, 64);
    softmaxT_kernel<<<dim3(256, 16), 256, 0, stream>>>(E, AT);
    renorm_kernel<<<dim3(256, 16), 256, 0, stream>>>(AT);
    // U = x - attn^T * V   (A = AT f32, B = V [n][v] plain)
    gemm16<EPI_SUBFROM, 1, 1, 1><<<g256, 32, 0, stream>>>(
        AT, 256, A256, Vh, 256, A256, Uh, 256, A256, nullptr, nullptr, nullptr,
        cur, 256, A256, 256, 256);
    // x_next = x + relu(bn(U * Wt^T + bt))
    gemm16<EPI_T_RES, 0, 1, 0><<<g256, 32, 0, stream>>>(
        Uh, 256, A256, wth[i], 256, 0, outs[i], 256, A256, saS[i], saT[i], bt,
        cur, 256, A256, 256, 256);
    cur = outs[i];
  }

  // fuse: F[b][n][0:1024] = [x1|x2|x3|x4|f1] (K blocks of 256) * fuse_w^T
  const _Float16* fins[5] = {x1h, x2h, x3h, x4h, f1h};
  for (int i = 0; i < 5; ++i) {
    if (i == 0)
      gemm16<EPI_NONE, 0, 0, 0><<<dim3(64, 16, 16), 32, 0, stream>>>(
          fins[i], 256, A256, fusewh + i * 256, 1280, 0, F, 1024, (size_t)262144,
          nullptr, nullptr, nullptr, nullptr, 0, 0, 1024, 256);
    else
      gemm16<EPI_ACCUM, 0, 0, 0><<<dim3(64, 16, 16), 32, 0, stream>>>(
          fins[i], 256, A256, fusewh + i * 256, 1280, 0, F, 1024, (size_t)262144,
          nullptr, nullptr, nullptr, nullptr, 0, 0, 1024, 256);
  }
  pool_kernel<<<64, 256, 0, stream>>>(F, fuseS, fuseT, pooled);

  // FC head (M=16 -> single WMMA row of tiles)
  gemm16<EPI_BN_LRELU, 1, 0, 0><<<dim3(32, 1, 1), 32, 0, stream>>>(
      pooled, 1024, 0, l1h, 1024, 0, h1b, 512, 0, bn6S, bn6T, nullptr,
      nullptr, 0, 0, 512, 1024);
  gemm16<EPI_BIAS_BN_LRELU, 1, 0, 0><<<dim3(16, 1, 1), 32, 0, stream>>>(
      h1b, 512, 0, l2h, 512, 0, h2b, 256, 0, bn7S, bn7T, F32(79),
      nullptr, 0, 0, 256, 512);
  gemm16<EPI_BIAS, 1, 0, 0><<<dim3(3, 1, 1), 32, 0, stream>>>(
      h2b, 256, 0, l3h, 256, 0, d_out, 40, 0, nullptr, nullptr, F32(85),
      nullptr, 0, 0, 40, 256);
}
